// involution_80942953660676
// MI455X (gfx1250) — matrix operand order repaired and checked by reading, not compiled, not run on hardware
//
#include <hip/hip_runtime.h>
#include <hip/hip_bf16.h>

// ---- problem constants -----------------------------------------------------
#define BATCH 4
#define CIN   256
#define HH    56
#define WW    56
#define PIX   (HH*WW)            // 3136
#define NTOT  (BATCH*PIX)        // 12544 = 98 * 128 exactly
#define CO    64                 // C/4
#define GRP   16
#define GCH   16
#define KK2   49
#define MW2   (GRP*KK2)          // 784 = 7 * 112 exactly
#define BN_EPS 1e-5f

// ---- workspace layout (floats). total = 10,637,568 f32 = ~42.6 MB ----------
#define T_OFF 0                          // t: [64][12544]
#define W_OFF (CO*NTOT)                  // per-pixel kernels: [784][12544]
#define S_OFF (W_OFF + MW2*NTOT)         // stats: sum[64] sq[64] scale[64] shift[64]

typedef __attribute__((ext_vector_type(16))) __bf16 v16bf;
typedef __attribute__((ext_vector_type(8)))  __bf16 v8bf;
typedef __attribute__((ext_vector_type(8)))  float  v8f;

// Build a 16-element bf16 fragment from two 16B-aligned contiguous LDS runs.
// Both the A fragment (run [kb,kb+16) of an [m][k] row) and the
// B fragment (run [kb,kb+16) of a [n][k]-order tile row) are contiguous in
// LDS, so each half is one aligned ds_load_b128.
__device__ __forceinline__ v16bf ld_frag(const __bf16* p0, const __bf16* p1) {
  v8bf lo = *(const v8bf*)p0;
  v8bf hi = *(const v8bf*)p1;
  return __builtin_shufflevector(lo, hi, 0, 1, 2, 3, 4, 5, 6, 7,
                                         8, 9, 10, 11, 12, 13, 14, 15);
}

// ---------------------------------------------------------------------------
__global__ void init_stats_kernel(float* __restrict__ p) {
  p[threadIdx.x] = 0.0f;   // zero sum[64] + sumsq[64]
}

// ---------------------------------------------------------------------------
// GEMM1: t[m][n] = sum_c w1[m][c] * x[c][n],  M=64, K=256, N=12544
// plus per-channel sum / sumsq for training-mode BatchNorm.
__global__ __launch_bounds__(256) void gemm1_stats_kernel(
    const float* __restrict__ x, const float* __restrict__ w1,
    float* __restrict__ t_ws, float* __restrict__ gsum, float* __restrict__ gsq)
{
  __shared__ __bf16 aw[CO][CIN + 8];     // all of w1 [m][k]; row = 528 B (16B mult)
  __shared__ __bf16 xt[128][40];         // x K-slab, fragment order [n][k]; row 80 B
  __shared__ float  lsum[CO], lsq[CO];

  const int tid  = threadIdx.x;
  const int lane = tid & 31;
  const int wv   = tid >> 5;     // wave 0..7 -> N-subtile
  const int hi   = lane >> 4;
  const int nl   = lane & 15;
  const int nloc = wv * 16 + nl; // this lane's column within the 128-wide tile
  const int ntile = blockIdx.x;  // 0..97

  if (tid < CO) { lsum[tid] = 0.0f; lsq[tid] = 0.0f; }

  // stage full w1 (64x256) as bf16; thread tid loads column tid of each row
  #pragma unroll 8
  for (int i = 0; i < CO; ++i)
    aw[i][tid] = (__bf16)w1[i * CIN + tid];

  v8f acc[4] = {};

  for (int k0 = 0; k0 < 8; ++k0) {       // K = 8 * 32
    __syncthreads();
    // load x tile, store transposed as [n][k] (global read coalesced along n)
    #pragma unroll
    for (int i = 0; i < 16; ++i) {
      int e   = tid + 256 * i;           // < 4096
      int nL  = e & 127;
      int row = e >> 7;                  // k within slab, 0..31
      int n   = ntile * 128 + nL;
      int b   = n / PIX;
      int p   = n - b * PIX;
      int c   = k0 * 32 + row;
      xt[nL][row] = (__bf16)x[(b * CIN + c) * PIX + p];
    }
    __syncthreads();

    // B fragment: one contiguous 16-value k-run per lane -> 2x ds_load_b128
    const int kbB = hi * 16;
    v16bf bf = ld_frag(&xt[nloc][kbB], &xt[nloc][kbB + 8]);

    const int kbA = k0 * 32 + hi * 16;
    #pragma unroll
    for (int mt = 0; mt < 4; ++mt) {
      int m = mt * 16 + nl;
      v16bf af = ld_frag(&aw[m][kbA], &aw[m][kbA + 8]);
      acc[mt] = __builtin_amdgcn_wmma_f32_16x16x32_bf16(
          false, af, false, bf, (short)0, acc[mt], false, false);
    }
  }
  __syncthreads();

  // epilogue: write t (n-contiguous) + LDS-atomic channel statistics
  const int nbase = ntile * 128 + nloc;
  #pragma unroll
  for (int mt = 0; mt < 4; ++mt) {
    #pragma unroll
    for (int v = 0; v < 8; ++v) {
      int   m   = mt * 16 + v + hi * 8;
      float val = acc[mt][v];
      t_ws[m * NTOT + nbase] = val;
      atomicAdd(&lsum[m], val);
      atomicAdd(&lsq[m],  val * val);
    }
  }
  __syncthreads();
  if (tid < CO) {
    atomicAdd(&gsum[tid], lsum[tid]);
    atomicAdd(&gsq[tid],  lsq[tid]);
  }
}

// ---------------------------------------------------------------------------
__global__ void bn_finalize_kernel(
    const float* __restrict__ gsum, const float* __restrict__ gsq,
    const float* __restrict__ gamma, const float* __restrict__ beta,
    float* __restrict__ scale, float* __restrict__ shift)
{
  int o = threadIdx.x;                   // 64 threads
  float inv_n = 1.0f / (float)NTOT;
  float mean  = gsum[o] * inv_n;
  float var   = gsq[o] * inv_n - mean * mean;   // biased, matches jnp.var
  float s     = gamma[o] * rsqrtf(var + BN_EPS);
  scale[o] = s;
  shift[o] = beta[o] - mean * s;
}

// ---------------------------------------------------------------------------
// GEMM2: weight[m][n] = sum_k w2[m][k] * relu(scale[k]*t[k][n] + shift[k])
// M=784 (7 blocks of 112), K=64, N=12544 (98 blocks of 128)
__global__ __launch_bounds__(256) void gemm2_kernel(
    const float* __restrict__ w2, const float* __restrict__ t_ws,
    const float* __restrict__ scale, const float* __restrict__ shift,
    float* __restrict__ weight_ws)
{
  __shared__ __bf16 a2[112][CO + 8];     // w2 M-block [m][k]; row 144 B (16B mult)
  __shared__ __bf16 ts[128][CO + 8];     // normalized t, fragment order [n][k]

  const int tid  = threadIdx.x;
  const int lane = tid & 31;
  const int wv   = tid >> 5;
  const int hi   = lane >> 4;
  const int nl   = lane & 15;
  const int nloc = wv * 16 + nl;
  const int m0    = blockIdx.x * 112;    // 0..6
  const int ntile = blockIdx.y;          // 0..97

  #pragma unroll
  for (int i = 0; i < 28; ++i) {         // 112*64 = 7168 elements
    int e  = tid + 256 * i;
    int k  = e & 63;
    int mm = e >> 6;
    a2[mm][k] = (__bf16)w2[(m0 + mm) * CO + k];
  }
  #pragma unroll
  for (int i = 0; i < 32; ++i) {         // 64*128 = 8192 elements
    int e = tid + 256 * i;
    int n = e & 127;
    int k = e >> 7;
    float v = t_ws[k * NTOT + ntile * 128 + n];
    v = v * scale[k] + shift[k];         // fused BatchNorm
    v = v > 0.0f ? v : 0.0f;             // fused ReLU
    ts[n][k] = (__bf16)v;                // store transposed -> fragment order
  }
  __syncthreads();

  v8f acc[7] = {};
  #pragma unroll
  for (int ks = 0; ks < 2; ++ks) {       // K = 2 * 32
    const int kbB = ks * 32 + hi * 16;
    v16bf bf = ld_frag(&ts[nloc][kbB], &ts[nloc][kbB + 8]);

    const int kbA = ks * 32 + hi * 16;
    #pragma unroll
    for (int mt = 0; mt < 7; ++mt) {
      int m = mt * 16 + nl;
      v16bf af = ld_frag(&a2[m][kbA], &a2[m][kbA + 8]);
      acc[mt] = __builtin_amdgcn_wmma_f32_16x16x32_bf16(
          false, af, false, bf, (short)0, acc[mt], false, false);
    }
  }

  const int nbase = ntile * 128 + nloc;
  #pragma unroll
  for (int mt = 0; mt < 7; ++mt)
    #pragma unroll
    for (int v = 0; v < 8; ++v)
      weight_ws[(m0 + mt * 16 + v + hi * 8) * NTOT + nbase] = acc[mt][v];
}

// ---------------------------------------------------------------------------
// Involution gather: out[b, g*16+gc, h, w] =
//   sum_{k2} weight[(g*49+k2)][n] * x[b, g*16+gc, h+di-3, w+dj-3]
// Block = (batch, group, 8-row strip). 16-channel padded halo staged in LDS;
// each dynamic weight loaded once and reused across all 16 group channels.
__global__ __launch_bounds__(256) void involution_kernel(
    const float* __restrict__ x, const float* __restrict__ weight_ws,
    float* __restrict__ out)
{
  __shared__ float xs[GCH][14][64];      // 57,344 B: 16 ch x (8+6) rows x padded cols

  const int tid = threadIdx.x;
  int bx = blockIdx.x;
  const int chunk = bx % 7;  bx /= 7;    // row strip
  const int g     = bx % GRP; bx /= GRP;
  const int b     = bx;                  // 0..3
  const int h0    = chunk * 8;

  // stage halo (zero padded): 16*14*64 = 14336 slots
  #pragma unroll 8
  for (int i = 0; i < 56; ++i) {
    int e   = tid + 256 * i;
    int col = e & 63;
    int r   = (e >> 6) % 14;
    int gc  = e / (64 * 14);
    int gr  = h0 - 3 + r;
    int gw  = col - 3;
    float v = 0.0f;
    if (gr >= 0 && gr < HH && gw >= 0 && gw < WW)
      v = x[((b * CIN + g * GCH + gc) * PIX) + gr * WW + gw];
    xs[gc][r][col] = v;
  }
  __syncthreads();

  for (int i = 0; i < 2; ++i) {
    int pix = tid + 256 * i;             // 448 pixels in strip
    if (pix < 8 * WW) {
      int hr = pix / WW;
      int wc = pix - hr * WW;
      int n  = b * PIX + (h0 + hr) * WW + wc;

      float accv[GCH];
      #pragma unroll
      for (int gc = 0; gc < GCH; ++gc) accv[gc] = 0.0f;

      for (int di = 0; di < 7; ++di) {
        #pragma unroll
        for (int dj = 0; dj < 7; ++dj) {
          int k2 = di * 7 + dj;
          float wvv = weight_ws[(g * KK2 + k2) * NTOT + n];  // coalesced in n
          #pragma unroll
          for (int gc = 0; gc < GCH; ++gc)
            accv[gc] += wvv * xs[gc][hr + di][wc + dj];
        }
      }
      #pragma unroll
      for (int gc = 0; gc < GCH; ++gc)
        out[((b * CIN + g * GCH + gc) * PIX) + (h0 + hr) * WW + wc] = accv[gc];
    }
  }
}

// ---------------------------------------------------------------------------
extern "C" void kernel_launch(void* const* d_in, const int* in_sizes, int n_in,
                              void* d_out, int out_size, void* d_ws, size_t ws_size,
                              hipStream_t stream)
{
  const float* x     = (const float*)d_in[0];  // [4,256,56,56]
  const float* w1    = (const float*)d_in[1];  // [64,256]
  const float* gamma = (const float*)d_in[2];  // [64]
  const float* beta  = (const float*)d_in[3];  // [64]
  const float* w2    = (const float*)d_in[4];  // [784,64]
  float* out = (float*)d_out;

  float* ws        = (float*)d_ws;             // needs ~42.6 MB
  float* t_ws      = ws + T_OFF;
  float* weight_ws = ws + W_OFF;
  float* gsum      = ws + S_OFF;
  float* gsq       = gsum + 64;
  float* scale     = gsum + 128;
  float* shift     = gsum + 192;

  init_stats_kernel<<<1, 128, 0, stream>>>(gsum);
  gemm1_stats_kernel<<<NTOT / 128, 256, 0, stream>>>(x, w1, t_ws, gsum, gsq);
  bn_finalize_kernel<<<1, 64, 0, stream>>>(gsum, gsq, gamma, beta, scale, shift);
  gemm2_kernel<<<dim3(MW2 / 112, NTOT / 128), 256, 0, stream>>>(
      w2, t_ws, scale, shift, weight_ws);
  involution_kernel<<<BATCH * GRP * (HH / 8), 256, 0, stream>>>(x, weight_ws, out);
}
